// GATLayer_3702261809871
// MI455X (gfx1250) — matrix operand order repaired and checked by reading, not compiled, not run on hardware
//
#include <hip/hip_runtime.h>
#include <hip/hip_bf16.h>
#include <cstddef>

#define NN 8192
#define IND 512
#define OUTD 64
#define ALPHA 0.2f

typedef __attribute__((ext_vector_type(2)))  float  v2f;
typedef __attribute__((ext_vector_type(8)))  float  v8f;
typedef __attribute__((ext_vector_type(4)))  int    v4i;
typedef __attribute__((ext_vector_type(8)))  __bf16 v8bf;
typedef __attribute__((ext_vector_type(16))) __bf16 v16bf;

// ---------------------------------------------------------------------------
// Kernel 1: h = X @ W  in exact fp32 via V_WMMA_F32_16X16X4_F32.
// Also emits hT = bf16(h)^T  ([OUTD][NN], row = output channel) so the
// attention pass can load WMMA B operands as contiguous 32B runs.
// One wave -> one 16x16 tile of h. 2048 tiles, 8 waves/block, 256 blocks.
// ---------------------------------------------------------------------------
__global__ __launch_bounds__(256) void gemm_h_kernel(
    const float* __restrict__ X, const float* __restrict__ W,
    float* __restrict__ h, __bf16* __restrict__ hT)
{
    const int wave = threadIdx.x >> 5;
    const int lane = threadIdx.x & 31;
    const int m    = lane & 15;        // row within tile (A) / col (B/C)
    const int half = lane >> 4;

    const int tile = blockIdx.x * 8 + wave;     // 0..2047
    const int i0   = (tile >> 2) << 4;          // 512 i-tiles
    const int d0   = (tile & 3) << 4;           // 4 d-tiles

    const float* xrow = X + (size_t)(i0 + m) * IND;
    const float* wcol = W + d0 + m;

    v8f acc = {};
    for (int k0 = 0; k0 < IND; k0 += 4) {
        const int k = k0 + 2 * half;
        // A 16x4 f32: lane<16 -> (K=k0,k0+1), lane>=16 -> (K=k0+2,k0+3)
        v2f a = *(const v2f*)(xrow + k);
        // B 4x16 f32: same K split across lane halves, N = lane&15
        v2f b;
        b.x = wcol[(size_t)k * OUTD];
        b.y = wcol[(size_t)(k + 1) * OUTD];
        acc = __builtin_amdgcn_wmma_f32_16x16x4_f32(
            false, a, false, b, (short)0, acc, false, false);
    }

    // C layout: VGPR r -> row (r + 8*half), col (lane&15).
    v8bf hv;
#pragma unroll
    for (int r = 0; r < 8; ++r) {
        const int row = i0 + 8 * half + r;       // consecutive in r
        h[(size_t)row * OUTD + d0 + m] = acc[r];
        hv[r] = (__bf16)acc[r];
    }
    *(v8bf*)(hT + (size_t)(d0 + m) * NN + i0 + 8 * half) = hv;
}

// ---------------------------------------------------------------------------
// Kernel 2: s1[i] = h[i,:] . a[0:64],  s2[i] = h[i,:] . a[64:128]
// ---------------------------------------------------------------------------
__global__ __launch_bounds__(256) void score_kernel(
    const float* __restrict__ h, const float* __restrict__ a,
    float* __restrict__ s1, float* __restrict__ s2)
{
    const int i = blockIdx.x * blockDim.x + threadIdx.x;
    if (i >= NN) return;
    const float* hr = h + (size_t)i * OUTD;
    float r1 = 0.f, r2 = 0.f;
#pragma unroll
    for (int d = 0; d < OUTD; ++d) {
        const float v = hr[d];
        r1 = fmaf(v, a[d], r1);
        r2 = fmaf(v, a[OUTD + d], r2);
    }
    s1[i] = r1;
    s2[i] = r2;
}

// ---------------------------------------------------------------------------
// Kernel 3: fused masked-softmax-weighted aggregation.
//   out[i,:] = (1/Z_i) * sum_j  e_ij * h[j,:],
//   e_ij = Adj[i,j] ? exp(lrelu(s1[i]+s2[j])) : 0,  Z_i = sum_j e_ij
// One wave per 16-row i-tile; four 16x16 f32 accumulators cover all 64 cols.
// j advances by 32 per step; E tile (16x32) built in registers as bf16 A,
// B operands streamed from bf16 hT. HBM cost = one NT pass over Adj.
// ---------------------------------------------------------------------------
__global__ __launch_bounds__(128) void gat_attn_kernel(
    const int* __restrict__ Adj,
    const float* __restrict__ s1, const float* __restrict__ s2,
    const __bf16* __restrict__ hT, float* __restrict__ out)
{
    __shared__ float s2s[NN];
    for (int t = threadIdx.x; t < NN; t += 128) s2s[t] = s2[t];
    __syncthreads();

    const int wave = threadIdx.x >> 5;
    const int lane = threadIdx.x & 31;
    const int m    = lane & 15;
    const int half = lane >> 4;

    const int i0 = (blockIdx.x * 4 + wave) << 4;
    const float s1v = s1[i0 + m];                      // A-row of this lane
    const size_t rowbase = (size_t)(i0 + m) * NN;

    v8f acc0 = {}, acc1 = {}, acc2 = {}, acc3 = {};
    float zsum = 0.f;

    for (int j0 = 0; j0 < NN; j0 += 32) {
        // --- Adj for this lane's 16 A slots: k_rel in {8h..8h+7} u {8h+16..8h+23}
        const int* ap = Adj + rowbase + j0 + 8 * half;
        const v4i A0 = __builtin_nontemporal_load((const v4i*)(ap));
        const v4i A1 = __builtin_nontemporal_load((const v4i*)(ap + 4));
        const v4i A2 = __builtin_nontemporal_load((const v4i*)(ap + 16));
        const v4i A3 = __builtin_nontemporal_load((const v4i*)(ap + 20));
        const int adjv[16] = {A0.x, A0.y, A0.z, A0.w, A1.x, A1.y, A1.z, A1.w,
                              A2.x, A2.y, A2.z, A2.w, A3.x, A3.y, A3.z, A3.w};

        float e[16];
#pragma unroll
        for (int q = 0; q < 16; ++q) {
            const int krel = 8 * half + (q < 8 ? q : q + 8);
            const float s = s1v + s2s[j0 + krel];
            const float l = s > 0.f ? s : ALPHA * s;
            const float ev = adjv[q] ? __expf(l) : 0.f;
            zsum += ev;
            e[q] = ev;
        }

        // A 16x32 bf16 layout: VGPR v, elt jj -> K = 8h + (v<4 ? 2v+jj : 16+2(v-4)+jj)
        v16bf A;
#pragma unroll
        for (int v = 0; v < 8; ++v) {
#pragma unroll
            for (int jj = 0; jj < 2; ++jj) {
                const int q = (v < 4) ? (2 * v + jj) : (8 + 2 * (v - 4) + jj);
                A[2 * v + jj] = (__bf16)e[q];
            }
        }

        // B 32x16 bf16: lane holds col n = lane&15, K = {8h..8h+7} u {16+8h..16+8h+7}
        const size_t boff = (size_t)j0 + 8 * half;
#define LDB(c) __builtin_shufflevector(*(const v8bf*)(hT + (size_t)((c) * 16 + m) * NN + boff), *(const v8bf*)(hT + (size_t)((c) * 16 + m) * NN + boff + 16), 0,1,2,3,4,5,6,7,8,9,10,11,12,13,14,15)
        const v16bf B0 = LDB(0);
        const v16bf B1 = LDB(1);
        const v16bf B2 = LDB(2);
        const v16bf B3 = LDB(3);
#undef LDB

        acc0 = __builtin_amdgcn_wmma_f32_16x16x32_bf16(false, A, false, B0, (short)0, acc0, false, false);
        acc1 = __builtin_amdgcn_wmma_f32_16x16x32_bf16(false, A, false, B1, (short)0, acc1, false, false);
        acc2 = __builtin_amdgcn_wmma_f32_16x16x32_bf16(false, A, false, B2, (short)0, acc2, false, false);
        acc3 = __builtin_amdgcn_wmma_f32_16x16x32_bf16(false, A, false, B3, (short)0, acc3, false, false);
    }

    // Full row sum: lane l and lane l^16 cover complementary K subsets of row m.
    zsum += __shfl_xor(zsum, 16, 32);

    // Normalize + store. C layout: VGPR r -> row (r + 8*half), col (lane&15).
    float Zr[8];
#pragma unroll
    for (int r = 0; r < 8; ++r)
        Zr[r] = __shfl(zsum, r + 8 * half, 32);   // Z of row (r+8h) lives in that lane

#pragma unroll
    for (int r = 0; r < 8; ++r) {
        const size_t ob = (size_t)(i0 + r + 8 * half) * OUTD + m;
        const float inv = 1.0f / Zr[r];
        out[ob + 0]  = acc0[r] * inv;
        out[ob + 16] = acc1[r] * inv;
        out[ob + 32] = acc2[r] * inv;
        out[ob + 48] = acc3[r] * inv;
    }
}

// ---------------------------------------------------------------------------
extern "C" void kernel_launch(void* const* d_in, const int* in_sizes, int n_in,
                              void* d_out, int out_size, void* d_ws, size_t ws_size,
                              hipStream_t stream) {
    (void)in_sizes; (void)n_in; (void)out_size; (void)ws_size;
    const float* X   = (const float*)d_in[0];
    const int*   Adj = (const int*)  d_in[1];
    const float* W   = (const float*)d_in[2];
    const float* a   = (const float*)d_in[3];
    float* out = (float*)d_out;

    char* ws = (char*)d_ws;
    float*  h  = (float*)ws;                                  // 2 MB
    __bf16* hT = (__bf16*)(ws + (size_t)2 * 1024 * 1024);     // 1 MB
    float*  s1 = (float*)(ws + (size_t)3 * 1024 * 1024);      // 32 KB
    float*  s2 = s1 + NN;                                     // 32 KB

    gemm_h_kernel<<<256, 256, 0, stream>>>(X, W, h, hT);
    score_kernel<<<NN / 256, 256, 0, stream>>>(h, a, s1, s2);
    gat_attn_kernel<<<NN / 64, 128, 0, stream>>>(Adj, s1, s2, hT, out);
}
